// RHSurrogate_1563368096576
// MI455X (gfx1250) — matrix operand order repaired and checked by reading, not compiled, not run on hardware
//
#include <hip/hip_runtime.h>
#include <hip/hip_bf16.h>

// ---------------------------------------------------------------------------
// Types for CDNA5 WMMA (wave32): bf16 16x16x32, f32 accumulate.
// ---------------------------------------------------------------------------
typedef __bf16 bf16;
typedef __attribute__((ext_vector_type(16))) __bf16 v16bf;
typedef __attribute__((ext_vector_type(8)))  float  v8f;

#define BB   2
#define CC   2048
#define DD   256
#define HH   8
#define DH   32
#define LL   4
#define FFN  2048
#define MM   (BB * CC)      // 4096 rows of activations
#define NOUT 16384

// ---------------------------------------------------------------------------
// Generic WMMA GEMM:  out[M,N] = A[M,K](bf16) * Bt[N,K](bf16)^T + bias
// optional: ReLU, residual add (f32 [M,N]), f32 out and/or bf16 out.
// Block: 128 threads (4 waves). C tile: 64(M) x 64(N). K step: 32.
// ---------------------------------------------------------------------------
__global__ __launch_bounds__(128)
void gemm_wmma_bf16(const bf16* __restrict__ A, const bf16* __restrict__ Bt,
                    const float* __restrict__ bias,
                    const float* __restrict__ residual,
                    float* __restrict__ outF, bf16* __restrict__ outH,
                    int M, int N, int K, int relu)
{
    __shared__ __align__(16) bf16 Asm[64 * 32];   // [row][k]
    __shared__ __align__(16) bf16 Bsm[64 * 32];   // [n][k] (already transposed)

    const int tid  = threadIdx.x;
    const int lane = tid & 31;
    const int wave = tid >> 5;
    const int bm   = blockIdx.y * 64;
    const int bn   = blockIdx.x * 64;

    v8f acc[4] = {};

    const int arow = (wave << 4) + (lane & 15);       // wave-local A row 0..63
    const int kbA  = (lane < 16) ? 0 : 16;            // A-frag K base (contiguous, matches B)
    const int kbB  = (lane < 16) ? 0 : 16;            // B-frag K base (contiguous, matches A)
    const int nl   = lane & 15;

    for (int k0 = 0; k0 < K; k0 += 32) {
        // Stage 64x32 bf16 tiles of A and Bt into LDS (16B vector moves).
#pragma unroll
        for (int t = 0; t < 2; ++t) {
            int idx = tid + t * 128;                  // 0..255
            int r   = idx >> 2;                       // tile row 0..63
            int c   = idx & 3;                        // 16B chunk 0..3
            uint4 va = *(const uint4*)(A  + (size_t)(bm + r) * K + k0 + c * 8);
            uint4 vb = *(const uint4*)(Bt + (size_t)(bn + r) * K + k0 + c * 8);
            *(uint4*)(&Asm[r * 32 + c * 8]) = va;
            *(uint4*)(&Bsm[r * 32 + c * 8]) = vb;
        }
        // Prefetch next K tile into the cache hierarchy (global_prefetch_b8).
        if (k0 + 32 < K) {
            __builtin_prefetch((const void*)(A  + (size_t)(bm + (tid >> 2)) * K + k0 + 32), 0, 1);
            __builtin_prefetch((const void*)(Bt + (size_t)(bn + (tid >> 2)) * K + k0 + 32), 0, 1);
        }
        __syncthreads();

        // A fragment: elems 0-15 -> K=kbA..kbA+15 (contiguous, same as B)
        v16bf a;
        {
            const bf16* ap = &Asm[arow * 32];
#pragma unroll
            for (int e = 0; e < 8; ++e) {
                a[e]     = ap[kbA + e];
                a[8 + e] = ap[kbA + 8 + e];
            }
        }
#pragma unroll
        for (int j = 0; j < 4; ++j) {
            v16bf b;
            const bf16* bp = &Bsm[(j * 16 + nl) * 32 + kbB];
#pragma unroll
            for (int e = 0; e < 16; ++e) b[e] = bp[e];
            acc[j] = __builtin_amdgcn_wmma_f32_16x16x32_bf16(
                false, a, false, b, (short)0, acc[j], false, false);
        }
        __syncthreads();
    }

    // Epilogue: C/D layout -> VGPR r: M = r + (lane<16 ? 0 : 8), N = (lane&15) + 16j
    const int mBase = bm + (wave << 4) + ((lane < 16) ? 0 : 8);
#pragma unroll
    for (int j = 0; j < 4; ++j) {
        int col  = bn + j * 16 + nl;
        float bv = bias ? bias[col] : 0.0f;
#pragma unroll
        for (int r = 0; r < 8; ++r) {
            int row = mBase + r;
            float v = acc[j][r] + bv;
            if (relu) v = fmaxf(v, 0.0f);
            if (residual) v += residual[(size_t)row * N + col];
            if (outF) outF[(size_t)row * N + col] = v;
            if (outH) outH[(size_t)row * N + col] = (bf16)v;
        }
    }
}

// ---------------------------------------------------------------------------
// LayerNorm over D=256: one wave per row, wave32 shuffle reduction, bf16 out.
// ---------------------------------------------------------------------------
__global__ __launch_bounds__(128)
void layernorm_bf16(const float* __restrict__ x, const float* __restrict__ g,
                    const float* __restrict__ b, bf16* __restrict__ out, int M)
{
    int wave = threadIdx.x >> 5, lane = threadIdx.x & 31;
    int row  = blockIdx.x * 4 + wave;
    if (row >= M) return;
    const float* xr = x + (size_t)row * DD;
    float4 v0 = ((const float4*)xr)[lane * 2];
    float4 v1 = ((const float4*)xr)[lane * 2 + 1];
    float vv[8] = { v0.x, v0.y, v0.z, v0.w, v1.x, v1.y, v1.z, v1.w };
    float s = 0.0f, sq = 0.0f;
#pragma unroll
    for (int e = 0; e < 8; ++e) { s += vv[e]; sq += vv[e] * vv[e]; }
#pragma unroll
    for (int off = 16; off; off >>= 1) {
        s  += __shfl_xor(s,  off);
        sq += __shfl_xor(sq, off);
    }
    float mu   = s * (1.0f / DD);
    float var  = sq * (1.0f / DD) - mu * mu;
    float rstd = rsqrtf(var + 1e-5f);
    int c0 = lane * 8;
#pragma unroll
    for (int e = 0; e < 8; ++e)
        out[(size_t)row * DD + c0 + e] = (bf16)((vv[e] - mu) * rstd * g[c0 + e] + b[c0 + e]);
}

// ---------------------------------------------------------------------------
// Input projection: x = tokens(4096x8) @ W_in(8x256) + b_in
// ---------------------------------------------------------------------------
__global__ void input_proj(const float* __restrict__ tok, const float* __restrict__ Win,
                           const float* __restrict__ bin, float* __restrict__ x, int total)
{
    int gid = blockIdx.x * blockDim.x + threadIdx.x;
    if (gid >= total) return;
    int r = gid >> 8, d = gid & 255;
    float s = bin[d];
#pragma unroll
    for (int j = 0; j < 8; ++j) s += tok[r * 8 + j] * Win[j * DD + d];
    x[gid] = s;
}

// Weight convert + transpose: wt[n*K+k] = bf16(w[k*N+n])
__global__ void conv_transpose_bf16(const float* __restrict__ w, bf16* __restrict__ wt,
                                    int K, int N)
{
    int gid = blockIdx.x * blockDim.x + threadIdx.x;
    if (gid >= K * N) return;
    int k = gid / N, n = gid - k * N;
    wt[(size_t)n * K + k] = (bf16)w[gid];
}

__global__ void to_bf16(const float* __restrict__ x, bf16* __restrict__ y, int n)
{
    int gid = blockIdx.x * blockDim.x + threadIdx.x;
    if (gid < n) y[gid] = (bf16)x[gid];
}

// ---------------------------------------------------------------------------
// RoPE + split heads:  qf/kf/vf [B*C, D]  ->  q/k/v [B*H, C, DH] (q,k roped)
// ---------------------------------------------------------------------------
__global__ void qkv_rope(const float* __restrict__ qf, const float* __restrict__ kf,
                         const float* __restrict__ vf,
                         float* __restrict__ q, float* __restrict__ k,
                         float* __restrict__ v, int total)
{
    int gid = blockIdx.x * blockDim.x + threadIdx.x;
    if (gid >= total) return;
    int d  = gid & 31;
    int i  = (gid >> 5) & (CC - 1);
    int bh = gid >> 16;                 // / (CC*32)
    int b  = bh >> 3, h = bh & 7;
    size_t rrow = ((size_t)b * CC + i) * DD + h * DH;
    size_t orow = ((size_t)bh * CC + i) * DH;

    v[orow + d] = vf[rrow + d];

    int   d0  = (d < 16) ? d : d - 16;
    float inv = __expf(-(float)d0 * (9.210340371976184f / 16.0f)); // 10000^(-d0/16)
    float ang = (float)i * inv;
    float cs = __cosf(ang), sn = __sinf(ang);
    float q1 = qf[rrow + d0], q2 = qf[rrow + d0 + 16];
    float k1 = kf[rrow + d0], k2 = kf[rrow + d0 + 16];
    if (d < 16) { q[orow + d] = q1 * cs - q2 * sn;  k[orow + d] = k1 * cs - k2 * sn; }
    else        { q[orow + d] = q1 * sn + q2 * cs;  k[orow + d] = k1 * sn + k2 * cs; }
}

// ---------------------------------------------------------------------------
// Banded attention, online softmax. One wave per (bh, query); lane = dim.
// allowed keys: j = i - t (mod C), t in [0, k_tokens)  — exactly the mask band.
// Writes merged-head bf16 [B*C, D] for the Wo GEMM.
// ---------------------------------------------------------------------------
__global__ __launch_bounds__(128)
void banded_attention(const float* __restrict__ q, const float* __restrict__ k,
                      const float* __restrict__ v, bf16* __restrict__ out,
                      const int* __restrict__ ktok_p)
{
    int wave = threadIdx.x >> 5, lane = threadIdx.x & 31;
    int gw   = blockIdx.x * 4 + wave;
    int bh   = gw >> 11;                // / CC
    int i    = gw & (CC - 1);
    int ktok = ktok_p[0];

    size_t base = (size_t)bh * CC * DH;
    float  qd   = q[base + (size_t)i * DH + lane] * 0.17677669529663687f; // 1/sqrt(32)

    float m = -1e30f, ssum = 0.0f, acc = 0.0f;
    for (int t = 0; t < ktok; ++t) {
        int j = i - t; if (j < 0) j += CC;
        size_t kr = base + (size_t)j * DH;
        float p = qd * k[kr + lane];
#pragma unroll
        for (int off = 16; off; off >>= 1) p += __shfl_xor(p, off);
        float mn = fmaxf(m, p);
        float fs = __expf(m - mn);
        float w  = __expf(p - mn);
        ssum = ssum * fs + w;
        acc  = acc  * fs + w * v[kr + lane];
        m = mn;
    }
    float o = acc / ssum;
    int b = bh >> 3, h = bh & 7;
    out[((size_t)b * CC + i) * DD + h * DH + lane] = (bf16)o;
}

// ---------------------------------------------------------------------------
// Host-side orchestration
// ---------------------------------------------------------------------------
extern "C" void kernel_launch(void* const* d_in, const int* in_sizes, int n_in,
                              void* d_out, int out_size, void* d_ws, size_t ws_size,
                              hipStream_t stream)
{
    const float* tok   = (const float*)d_in[0];
    const float* W_in  = (const float*)d_in[1];
    const float* b_in  = (const float*)d_in[2];
    const float* Wq    = (const float*)d_in[3];
    const float* bq    = (const float*)d_in[4];
    const float* Wk    = (const float*)d_in[5];
    const float* bk    = (const float*)d_in[6];
    const float* Wv    = (const float*)d_in[7];
    const float* bv    = (const float*)d_in[8];
    const float* Wo    = (const float*)d_in[9];
    const float* bo    = (const float*)d_in[10];
    const float* ln1_g = (const float*)d_in[11];
    const float* ln1_b = (const float*)d_in[12];
    const float* ln2_g = (const float*)d_in[13];
    const float* ln2_b = (const float*)d_in[14];
    const float* W1    = (const float*)d_in[15];
    const float* b1    = (const float*)d_in[16];
    const float* W2    = (const float*)d_in[17];
    const float* b2    = (const float*)d_in[18];
    const float* W_out = (const float*)d_in[19];
    const float* b_out = (const float*)d_in[20];
    const int*   ktok  = (const int*)d_in[21];
    float* out = (float*)d_out;

    // Workspace carve-out (256B aligned).
    char* wsp = (char*)d_ws;
    auto alloc = [&](size_t bytes) -> char* {
        char* p = wsp;
        wsp += (bytes + 255) & ~(size_t)255;
        return p;
    };
    float* X    = (float*)alloc((size_t)MM * DD * 4);       // running activations
    bf16*  HB   = (bf16*) alloc((size_t)MM * DD * 2);       // LN output (GEMM A)
    float* QF   = (float*)alloc((size_t)MM * DD * 4);
    float* KF   = (float*)alloc((size_t)MM * DD * 4);
    float* VF   = (float*)alloc((size_t)MM * DD * 4);
    float* Qh   = (float*)alloc((size_t)MM * DD * 4);       // [B*H, C, DH]
    float* Kh   = (float*)alloc((size_t)MM * DD * 4);
    float* Vh   = (float*)alloc((size_t)MM * DD * 4);
    bf16*  ATT  = (bf16*) alloc((size_t)MM * DD * 2);       // merged-head attn out
    bf16*  FFB  = (bf16*) alloc((size_t)MM * FFN * 2);      // relu(h2@W1+b1)
    bf16*  XB   = (bf16*) alloc((size_t)MM * DD * 2);       // bf16(x) for final GEMM
    bf16*  WQT  = (bf16*) alloc((size_t)LL * DD * DD * 2);  // transposed bf16 weights
    bf16*  WKT  = (bf16*) alloc((size_t)LL * DD * DD * 2);
    bf16*  WVT  = (bf16*) alloc((size_t)LL * DD * DD * 2);
    bf16*  WOT  = (bf16*) alloc((size_t)LL * DD * DD * 2);
    bf16*  W1T  = (bf16*) alloc((size_t)LL * DD * FFN * 2);
    bf16*  W2T  = (bf16*) alloc((size_t)LL * FFN * DD * 2);
    bf16*  WOutT= (bf16*) alloc((size_t)DD * NOUT * 2);

    auto cvt = [&](const float* w, bf16* wt, int K, int N) {
        int n = K * N;
        conv_transpose_bf16<<<(n + 255) / 256, 256, 0, stream>>>(w, wt, K, N);
    };
    for (int l = 0; l < LL; ++l) {
        cvt(Wq + (size_t)l * DD * DD,  WQT + (size_t)l * DD * DD,  DD, DD);
        cvt(Wk + (size_t)l * DD * DD,  WKT + (size_t)l * DD * DD,  DD, DD);
        cvt(Wv + (size_t)l * DD * DD,  WVT + (size_t)l * DD * DD,  DD, DD);
        cvt(Wo + (size_t)l * DD * DD,  WOT + (size_t)l * DD * DD,  DD, DD);
        cvt(W1 + (size_t)l * DD * FFN, W1T + (size_t)l * DD * FFN, DD, FFN);
        cvt(W2 + (size_t)l * FFN * DD, W2T + (size_t)l * FFN * DD, FFN, DD);
    }
    cvt(W_out, WOutT, DD, NOUT);

    auto gemm = [&](const bf16* A, const bf16* Bt, const float* bias,
                    const float* residual, float* oF, bf16* oH,
                    int M, int N, int K, int relu) {
        dim3 grid(N / 64, M / 64);
        gemm_wmma_bf16<<<grid, 128, 0, stream>>>(A, Bt, bias, residual, oF, oH,
                                                 M, N, K, relu);
    };

    // x = tokens @ W_in + b_in
    input_proj<<<(MM * DD) / 256, 256, 0, stream>>>(tok, W_in, b_in, X, MM * DD);

    const int ropeTotal = BB * HH * CC * DH;     // 2^20
    for (int l = 0; l < LL; ++l) {
        const size_t wO = (size_t)l * DD * DD;
        // h = LN1(x)
        layernorm_bf16<<<MM / 4, 128, 0, stream>>>(X, ln1_g + l * DD, ln1_b + l * DD, HB, MM);
        // q,k,v projections
        gemm(HB, WQT + wO, bq + l * DD, nullptr, QF, nullptr, MM, DD, DD, 0);
        gemm(HB, WKT + wO, bk + l * DD, nullptr, KF, nullptr, MM, DD, DD, 0);
        gemm(HB, WVT + wO, bv + l * DD, nullptr, VF, nullptr, MM, DD, DD, 0);
        // RoPE + head layout
        qkv_rope<<<ropeTotal / 256, 256, 0, stream>>>(QF, KF, VF, Qh, Kh, Vh, ropeTotal);
        // banded attention (exactly equivalent to the dense masked softmax)
        banded_attention<<<(BB * HH * CC) / 4, 128, 0, stream>>>(Qh, Kh, Vh, ATT, ktok);
        // x = x + attn @ Wo + bo
        gemm(ATT, WOT + wO, bo + l * DD, X, X, nullptr, MM, DD, DD, 0);
        // h2 = LN2(x)
        layernorm_bf16<<<MM / 4, 128, 0, stream>>>(X, ln2_g + l * DD, ln2_b + l * DD, HB, MM);
        // f = relu(h2 @ W1 + b1)   (bf16 out feeds next GEMM)
        gemm(HB, W1T + (size_t)l * DD * FFN, b1 + l * FFN, nullptr, nullptr, FFB,
             MM, FFN, DD, 1);
        // x = x + f @ W2 + b2
        gemm(FFB, W2T + (size_t)l * FFN * DD, b2 + l * DD, X, X, nullptr, MM, DD, FFN, 0);
    }

    // out = x @ W_out + b_out
    to_bf16<<<(MM * DD) / 256, 256, 0, stream>>>(X, XB, MM * DD);
    gemm(XB, WOutT, b_out, nullptr, out, nullptr, MM, NOUT, DD, 0);
}
